// DeepLagrangianNetwork_14551349198985
// MI455X (gfx1250) — matrix operand order repaired and checked by reading, not compiled, not run on hardware
//
#include <hip/hip_runtime.h>
#include <hip/hip_bf16.h>
#include <stdint.h>

// ---------------------------------------------------------------------------
// DeepLagrangianNetwork fused forward for MI455X (gfx1250, wave32, WMMA).
//
// Each batch element = 8-row block (activation + 7 forward-mode tangents)
// pushed through the MLP as bf16 WMMA GEMMs with fp32 accumulation.
//
//  * Block = 4 batch elements = 32 rows = two 16-row M-tiles in LDS
//    (2 x ~64.5KB bf16 ping-pong slabs; CDNA5 allows 320KB per workgroup).
//  * Register blocking: kt outer; each wave keeps 2 M-tiles x 8 n-tiles of
//    f32 accumulators (128 VGPRs) live and shares each B fragment between
//    both M-tiles -> dense back-to-back v_wmma issue, 1 B-load pair / 2 wmma.
//  * LDS row stride padded to 1032 bf16 (516 dwords, 516 % 64 = 4): the 16
//    lanes of an A-fragment ds_load_b128 start at banks 4m -> all 64 banks
//    covered exactly once -> conflict-free (stride 1024 would be 16-way).
//
// WMMA fragment layout assumptions (compile-only, documented):
//   A (16x32 bf16): ISA 7.12.2 -- lanes 0-15: M=lane, K {0..7,16..23};
//                   lanes 16-31: M=lane-16, K {8..15,24..31}.
//   B (32x16 bf16): packed so N = lane%16, 16 bf16 = K in table order (32B/lane).
//   D (16x16 f32) : lanes 0-15: N=lane, M=vgpr; lanes 16-31: N=lane-16,
//                   M=vgpr+8  => row-0 preactivation is acc[0] of the SAME
//                   lane as its tangent rows -> in-lane ReLU gating.
// ---------------------------------------------------------------------------

#define BATCH   8192
#define NDOF    7
#define WID     1024
#define ASTRIDE (WID + 8)     // LDS row stride in bf16: bank-conflict padding
#define EPSR    1e-5f

typedef __attribute__((ext_vector_type(16))) __bf16 v16bf;
typedef __attribute__((ext_vector_type(8)))  __bf16 v8bf;
typedef __attribute__((ext_vector_type(8)))  float  v8f;

__device__ __forceinline__ uint16_t f2bf(float f) {
  uint32_t x = __float_as_uint(f);
  uint32_t r = (x + 0x7FFFu + ((x >> 16) & 1u)) >> 16;   // round-nearest-even
  return (uint16_t)r;
}

// --------------------------- weight pre-pack -------------------------------
// dst tile order: [nt][kt][lane(0..31)][s(0..15)], n = lane%16, k = WMMA table order, all bf16.
__global__ void pack_kxn(const float* __restrict__ src, uint16_t* __restrict__ dst,
                         int Ksrc, int nkt) {
  int idx = blockIdx.x * blockDim.x + threadIdx.x;
  int total = 64 * nkt * 32 * 16;                 // 64 n-tiles (N = 1024)
  if (idx >= total) return;
  int s  = idx & 15;
  int l  = (idx >> 4) & 31;
  int n  = l & 15;
  int k  = (s < 8) ? ((l >> 4) << 3) + s : 16 + ((l >> 4) << 3) + (s - 8);
  int c  = idx >> 9;                              // nt*nkt + kt
  int kt = c % nkt;
  int nt = c / nkt;
  int kg = kt * 32 + k;
  int ng = nt * 16 + n;
  float v = (kg < Ksrc) ? src[(size_t)kg * 1024 + ng] : 0.f;
  dst[idx] = f2bf(v);
}

// Heads: cols 0 = W_g, 1..7 = W_ld, 8..28 = W_lo, 29..31 = 0. N = 32.
__global__ void pack_heads(const float* __restrict__ Wg, const float* __restrict__ Wld,
                           const float* __restrict__ Wlo, uint16_t* __restrict__ dst) {
  int idx = blockIdx.x * blockDim.x + threadIdx.x;
  if (idx >= 2 * 32 * 32 * 16) return;
  int s  = idx & 15;
  int l  = (idx >> 4) & 31;
  int n  = l & 15;
  int k  = (s < 8) ? ((l >> 4) << 3) + s : 16 + ((l >> 4) << 3) + (s - 8);
  int kt = (idx >> 9) & 31;
  int nt = idx >> 14;
  int kg  = kt * 32 + k;
  int col = nt * 16 + n;
  float v;
  if      (col == 0) v = Wg[kg];
  else if (col < 8)  v = Wld[(size_t)kg * 7  + (col - 1)];
  else if (col < 29) v = Wlo[(size_t)kg * 21 + (col - 8)];
  else               v = 0.f;
  dst[idx] = f2bf(v);
}

// ------------------------------ fragments ----------------------------------
__device__ __forceinline__ v16bf load_afrag(const uint16_t* A, int kt, int lane) {
  int m  = lane & 15;
  int kh = (lane >> 4) << 3;                      // 0 or 8
  const uint16_t* p = A + m * ASTRIDE + kt * 32 + kh;
  union { v16bf v; v8bf h[2]; } u;
  u.h[0] = *(const v8bf*)p;                       // K +0..7   (ds_load_b128)
  u.h[1] = *(const v8bf*)(p + 16);                // K +16..23 (ds_load_b128)
  return u.v;
}

__device__ __forceinline__ v16bf load_bfrag(const uint16_t* Wp, int nt, int kt,
                                            int nkt, int lane) {
  // 32 contiguous bytes per lane; whole wave reads one 1KB tile (L2-hot).
  return *(const v16bf*)(Wp + (((size_t)(nt * nkt + kt) * 32 + lane) << 4));
}

// -------------------------- fused MLP layer --------------------------------
// A, O: 32 x ASTRIDE bf16 LDS slabs (rows 0-15 = M-tile 0, 16-31 = M-tile 1).
// Each wave owns a 256-col N strip, done in two passes of 8 n-tiles so the
// 2x8 accumulator block (128 VGPRs) stays in registers.
__device__ __forceinline__ void mlp_layer(const uint16_t* A, uint16_t* O,
                                          const uint16_t* Wp, const float* bias,
                                          int nkt, int lane, int wave) {
  for (int pass = 0; pass < 2; ++pass) {
    int nt0 = wave * 16 + pass * 8;
    v8f acc[2][8] = {};
    for (int kt = 0; kt < nkt; ++kt) {
      v16bf a0 = load_afrag(A,                kt, lane);   // M-tile 0
      v16bf a1 = load_afrag(A + 16 * ASTRIDE, kt, lane);   // M-tile 1
#pragma unroll
      for (int t = 0; t < 8; ++t) {
        v16bf b = load_bfrag(Wp, nt0 + t, kt, nkt, lane);
        acc[0][t] = __builtin_amdgcn_wmma_f32_16x16x32_bf16(
            false, a0, false, b, (short)0, acc[0][t], false, false);
        acc[1][t] = __builtin_amdgcn_wmma_f32_16x16x32_bf16(
            false, a1, false, b, (short)0, acc[1][t], false, false);
      }
    }
    // ReLU + tangent gating, all in-lane (see D layout note above).
#pragma unroll
    for (int mt = 0; mt < 2; ++mt) {
#pragma unroll
      for (int t = 0; t < 8; ++t) {
        int ncol  = (nt0 + t) * 16 + (lane & 15);
        int rbase = mt * 16 + ((lane >> 4) << 3);
        float a0v = acc[mt][t][0] + bias[ncol];
        float g   = (a0v > 0.f) ? 1.f : 0.f;
        O[(rbase + 0) * ASTRIDE + ncol] = f2bf(fmaxf(a0v, 0.f));
#pragma unroll
        for (int v = 1; v < 8; ++v)
          O[(rbase + v) * ASTRIDE + ncol] = f2bf(g * acc[mt][t][v]);
      }
    }
  }
}

__device__ __forceinline__ void head_layer(const uint16_t* A, const uint16_t* Wp,
                                           const float* bg, const float* bld,
                                           const float* blo, float* hb,
                                           int lane, int wave) {
  if (wave >= 2) return;                          // heads are only 32 cols wide
  int nt = wave;
  v8f acc[2] = {};
  for (int kt = 0; kt < 32; ++kt) {
    v16bf a0 = load_afrag(A,                kt, lane);
    v16bf a1 = load_afrag(A + 16 * ASTRIDE, kt, lane);
    v16bf b  = load_bfrag(Wp, nt, kt, 32, lane);
    acc[0] = __builtin_amdgcn_wmma_f32_16x16x32_bf16(
        false, a0, false, b, (short)0, acc[0], false, false);
    acc[1] = __builtin_amdgcn_wmma_f32_16x16x32_bf16(
        false, a1, false, b, (short)0, acc[1], false, false);
  }
  int n = nt * 16 + (lane & 15);
  float bias; int mode;                           // 0 = linear, 1 = relu-gated
  if      (n == 0) { bias = bg[0];       mode = 0; }   // V head
  else if (n < 8)  { bias = bld[n - 1];  mode = 1; }   // l_diag (relu)
  else if (n < 29) { bias = blo[n - 8];  mode = 0; }   // l_low (linear)
  else return;                                         // pad cols: never read
#pragma unroll
  for (int mt = 0; mt < 2; ++mt) {
    float* o = hb + (mt * 2 + (lane >> 4)) * 256;      // batch within 4-group
    float a0v = acc[mt][0] + bias;
    if (mode == 1) {
      float g = (a0v > 0.f) ? 1.f : 0.f;
      o[n] = fmaxf(a0v, 0.f);
#pragma unroll
      for (int v = 1; v < 8; ++v) o[v * 32 + n] = g * acc[mt][v];
    } else {
      o[n] = a0v;
#pragma unroll
      for (int v = 1; v < 8; ++v) o[v * 32 + n] = acc[mt][v];
    }
  }
}

// --------------------------- fused MLP kernel ------------------------------
__global__ __launch_bounds__(128) void fused_mlp(
    const float* __restrict__ state,
    const uint16_t* __restrict__ W0p, const uint16_t* __restrict__ W1p,
    const uint16_t* __restrict__ WHp,
    const float* __restrict__ b_in, const float* __restrict__ b_h,
    const float* __restrict__ b_g, const float* __restrict__ b_ld,
    const float* __restrict__ b_lo,
    float* __restrict__ headbuf) {
  __shared__ __align__(16) uint16_t sA[32 * ASTRIDE];   // ~64.5 KB
  __shared__ __align__(16) uint16_t sB[32 * ASTRIDE];   // ~64.5 KB (ping-pong)
  int tid = threadIdx.x, lane = tid & 31, wave = tid >> 5;
  int b0 = blockIdx.x * 4;                          // 4 batch elements / block

  // Layer-0 input block: row 0 = [state, 0...], row d = e_{d-1} (tangent seed).
  for (int e = tid; e < 32 * 32; e += 128) {
    int row = e >> 5, k = e & 31;
    int batch = b0 + (row >> 3), sub = row & 7;
    float v;
    if (sub == 0) v = (k < NDOF) ? state[(size_t)batch * NDOF + k] : 0.f;
    else          v = (k == sub - 1) ? 1.f : 0.f;
    sA[row * ASTRIDE + k] = f2bf(v);
  }
  __syncthreads();
  mlp_layer(sA, sB, W0p, b_in, 1, lane, wave);      // 7->1024 (K padded to 32)
  __syncthreads();
  mlp_layer(sB, sA, W1p, b_h, 32, lane, wave);      // 1024->1024
  __syncthreads();
  mlp_layer(sA, sB, W1p, b_h, 32, lane, wave);      // 1024->1024
  __syncthreads();
  head_layer(sB, WHp, b_g, b_ld, b_lo, headbuf + (size_t)b0 * 256, lane, wave);
}

// ----------------------- rigid-body epilogue -------------------------------
__global__ void dyn_epilogue(const float* __restrict__ head,
                             const float* __restrict__ vel,
                             const float* __restrict__ acc,
                             float* __restrict__ out) {
  int b = blockIdx.x * blockDim.x + threadIdx.x;
  if (b >= BATCH) return;
  const int R[28]  = {0,1,2,3,4,5,6, 1,2,3,4,5,6, 2,3,4,5,6, 3,4,5,6, 4,5,6, 5,6, 6};
  const int Cc[28] = {0,1,2,3,4,5,6, 0,1,2,3,4,5, 0,1,2,3,4, 0,1,2,3, 0,1,2, 0,1, 0};
  const float* h = head + (size_t)b * 256;          // h[row*32 + col]
  float v[NDOF], aq[NDOF];
#pragma unroll
  for (int i = 0; i < NDOF; ++i) { v[i] = vel[(size_t)b*NDOF+i]; aq[i] = acc[(size_t)b*NDOF+i]; }

  float L[NDOF][NDOF] = {};
#pragma unroll
  for (int t = 0; t < 28; ++t) L[R[t]][Cc[t]] = h[1 + t];

  float M[NDOF][NDOF];
#pragma unroll
  for (int i = 0; i < NDOF; ++i)
#pragma unroll
    for (int j = 0; j < NDOF; ++j) {
      float s = 0.f;
#pragma unroll
      for (int m = 0; m < NDOF; ++m) s += L[i][m] * L[j][m];
      M[i][j] = s + ((i == j) ? EPSR : 0.f);
    }

  float u[NDOF];                                    // u = L^T v
#pragma unroll
  for (int c = 0; c < NDOF; ++c) {
    float s = 0.f;
#pragma unroll
    for (int r = 0; r < NDOF; ++r) s += L[r][c] * v[r];
    u[c] = s;
  }

  float dldt[28];
#pragma unroll
  for (int t = 0; t < 28; ++t) {
    float s = 0.f;
#pragma unroll
    for (int d = 0; d < NDOF; ++d) s += h[(d + 1) * 32 + 1 + t] * v[d];
    dldt[t] = s;
  }

  // dMdt*v = L*(dLdt^T v) + dLdt*(L^T v)
  float w[NDOF] = {}, p[NDOF] = {};
#pragma unroll
  for (int t = 0; t < 28; ++t) {
    w[Cc[t]] += dldt[t] * v[R[t]];
    p[R[t]]  += dldt[t] * u[Cc[t]];
  }
  float Cv[NDOF];
#pragma unroll
  for (int i = 0; i < NDOF; ++i) {
    float s = 0.f;
#pragma unroll
    for (int m = 0; m < NDOF; ++m) s += L[i][m] * w[m];
    Cv[i] = s + p[i];
  }
  // quad_k = v^T(dLdq_k L^T + L dLdq_k^T)v = 2*(v^T dLdq_k)·u ; C -= 0.5*quad
#pragma unroll
  for (int k = 0; k < NDOF; ++k) {
    float s = 0.f;
#pragma unroll
    for (int t = 0; t < 28; ++t) s += v[R[t]] * h[(k + 1) * 32 + 1 + t] * u[Cc[t]];
    Cv[k] -= s;
  }
  float G[NDOF];
#pragma unroll
  for (int i = 0; i < NDOF; ++i) G[i] = h[(i + 1) * 32];

  float* tauO = out + (size_t)b * 7;
  float* MO   = out + 57344  + (size_t)b * 49;      // 8192*7
  float* CO   = out + 458752 + (size_t)b * 7;       // 8192*(7+49)
  float* GO   = out + 516096 + (size_t)b * 7;       // 8192*(7+49+7)
#pragma unroll
  for (int i = 0; i < NDOF; ++i) {
    float s = 0.f;
#pragma unroll
    for (int j = 0; j < NDOF; ++j) s += M[i][j] * aq[j];
    tauO[i] = s + Cv[i] + G[i];
    CO[i]   = Cv[i];
    GO[i]   = G[i];
#pragma unroll
    for (int j = 0; j < NDOF; ++j) MO[i * 7 + j] = M[i][j];
  }
}

// ------------------------------- launch ------------------------------------
extern "C" void kernel_launch(void* const* d_in, const int* in_sizes, int n_in,
                              void* d_out, int out_size, void* d_ws, size_t ws_size,
                              hipStream_t stream) {
  const float* state = (const float*)d_in[0];
  const float* vel   = (const float*)d_in[1];
  const float* acc   = (const float*)d_in[2];
  const float* W_in  = (const float*)d_in[3];
  const float* b_in  = (const float*)d_in[4];
  const float* W_h   = (const float*)d_in[5];
  const float* b_h   = (const float*)d_in[6];
  const float* W_g   = (const float*)d_in[7];
  const float* b_g   = (const float*)d_in[8];
  const float* W_ld  = (const float*)d_in[9];
  const float* b_ld  = (const float*)d_in[10];
  const float* W_lo  = (const float*)d_in[11];
  const float* b_lo  = (const float*)d_in[12];
  float* out = (float*)d_out;

  char* ws = (char*)d_ws;
  uint16_t* W0p     = (uint16_t*)(ws);                       //  64 KB (32x1024)
  uint16_t* W1p     = (uint16_t*)(ws + 65536);               //   2 MB (1024x1024)
  uint16_t* WHp     = (uint16_t*)(ws + 65536 + 2097152);     //  64 KB (1024x32)
  float*    headbuf = (float*)   (ws + 65536 + 2097152 + 65536); // 8 MB (8192x8x32)

  // 1) weight repack fp32 -> bf16, WMMA tile layout
  pack_kxn<<<(64 * 1 * 512 + 255) / 256, 256, 0, stream>>>(W_in, W0p, 7, 1);
  pack_kxn<<<(64 * 32 * 512 + 255) / 256, 256, 0, stream>>>(W_h, W1p, 1024, 32);
  pack_heads<<<(32768 + 255) / 256, 256, 0, stream>>>(W_g, W_ld, W_lo, WHp);

  // 2) fused 3-layer MLP + heads: 1 block = 4 batch elements (32 WMMA rows)
  fused_mlp<<<BATCH / 4, 128, 0, stream>>>(state, W0p, W1p, WHp,
                                           b_in, b_h, b_g, b_ld, b_lo, headbuf);

  // 3) rigid-body dynamics assembly (tau, M, C, G)
  dyn_epilogue<<<BATCH / 256, 256, 0, stream>>>(headbuf, vel, acc, out);
}
